// SlayerMLP_53291954209113
// MI455X (gfx1250) — matrix operand order repaired and checked by reading, not compiled, not run on hardware
//
#include <hip/hip_runtime.h>
#include <math.h>

typedef __attribute__((ext_vector_type(16))) _Float16 v16h;
typedef __attribute__((ext_vector_type(8)))  float    v8f;
typedef __attribute__((ext_vector_type(4)))  _Float16 h4;

#define B_SZ   32
#define NIN    2048
#define NOUT   1024
#define T_SMP  500
#define THETA  10.0f

#define TILE_M 128
#define TILE_N 128
#define TILE_K 64                    // two 32-deep k-halves per LDS stage
#define KSTEPS (NIN / TILE_K)        // 32

struct SrmTaps { float v[96]; int n; };
struct RefTaps { float v[16]; };

// ---------------------------------------------------------------------------
// Kernel 1: Z_b(1024 x 500) = W(1024 x 2048) x S_b(2048 x 500), f16 WMMA.
// 256 threads (8 waves), tile 128(M) x 128(N), K-stage 64 (2 k-halves),
// LDS double-buffered (64 KB), fragments stored pre-swizzled in the exact
// v_wmma_f32_16x16x32_f16 layout (32 contiguous bytes per lane).
// GUARD=false: all 128 N columns in range (no masks, no col checks).
// ---------------------------------------------------------------------------
template<bool GUARD>
__global__ __launch_bounds__(256)
void slayer_gemm_wmma(const float* __restrict__ S,   // (B, NIN, T)
                      const float* __restrict__ W,   // (NOUT, NIN)
                      float* __restrict__ Z,         // (B, NOUT, T)
                      int nBase0)
{
  // per k-half: A = 8 M-subtiles (16x32), B = 8 N-subtiles (32x16), 512 halfs
  __shared__ __align__(32) _Float16 Abuf[2][2 * 4096];
  __shared__ __align__(32) _Float16 Bbuf[2][2 * 4096];

  const int tid   = threadIdx.x;
  const int lane  = tid & 31;
  const int wave  = tid >> 5;          // 0..7
  const int waveM = wave & 3;          // M subtiles 2*waveM, 2*waveM+1
  const int waveN = wave >> 2;         // N subtiles 4*waveN .. 4*waveN+3

  const int mBase = blockIdx.y * TILE_M;
  const int nBase = nBase0 + blockIdx.x * TILE_N;
  const int b     = blockIdx.z;

  const float* __restrict__ Wm = W + (size_t)mBase * NIN;
  const float* __restrict__ Sb = S + (size_t)b * NIN * T_SMP + nBase;

  // ---- A staging: group i: row = tid>>4 + 16*i (subtile == i), kq = (tid&15)*4
  const int aRow0  = tid >> 4;                    // 0..15  (row & 15)
  const int aKq    = (tid & 15) * 4;              // 0..60
  const int aKhalf = aKq >> 5;
  const int aKk    = aKq & 31;
  const int aOff0  = aRow0 * NIN + aKq;           // + i*16*NIN
  const int aLn    = aRow0 + 16 * ((aKk >> 3) & 1);
  const int aE0    = (aKk & 7) + 8 * ((aKk >> 4) & 1);
  const int aDst0  = aKhalf * 4096 + aLn * 16 + aE0;   // + i*512

  // ---- B staging: group i: kk = tid>>5 + 8*i (khalf = i>>2), nq = (tid&31)*4
  const int bKk0  = tid >> 5;                     // 0..7
  const int bNq   = (tid & 31) * 4;               // 0..124
  const int bInRange = GUARD ? (nBase + bNq < T_SMP) : 1;
  const int bOffSafe = bInRange ? (bKk0 * T_SMP + bNq) : bKk0 * T_SMP; // clamp
  const int bDstN = (bNq >> 4) * 512 + (bNq & 15) * 16;

  auto stage = [&](int ks, int buf) {
    const float* wk = Wm + ks * TILE_K;                        // uniform
    const float* sk = Sb + (size_t)(ks * TILE_K) * T_SMP;      // uniform
    _Float16* Ad = &Abuf[buf][0];
    _Float16* Bd = &Bbuf[buf][0];
#pragma unroll
    for (int i = 0; i < 8; ++i) {
      float4 v = *(const float4*)(wk + (aOff0 + i * 16 * NIN));
      h4 h;
      h.x = (_Float16)v.x; h.y = (_Float16)v.y;
      h.z = (_Float16)v.z; h.w = (_Float16)v.w;
      *(h4*)(Ad + (aDst0 + i * 512)) = h;                      // ds_store_b64
    }
#pragma unroll
    for (int i = 0; i < 8; ++i) {
      const int kk  = bKk0 + 8 * (i & 3);                      // 0..31 in half
      const int kh  = i >> 2;                                  // k-half
      float4 v = *(const float4*)(sk + (bOffSafe + i * 8 * T_SMP));
      if (GUARD && !bInRange) v = make_float4(0.f, 0.f, 0.f, 0.f);
      const int base = kh * 4096 + bDstN + ((kk >> 3) & 1) * 256 + (kk & 7) + 8 * (kk >> 4);
      Bd[base +  0] = (_Float16)v.x;
      Bd[base + 16] = (_Float16)v.y;
      Bd[base + 32] = (_Float16)v.z;
      Bd[base + 48] = (_Float16)v.w;
    }
  };

  v8f zero = {0.f, 0.f, 0.f, 0.f, 0.f, 0.f, 0.f, 0.f};
  v8f acc[2][4];
#pragma unroll
  for (int i = 0; i < 2; ++i)
#pragma unroll
    for (int j = 0; j < 4; ++j) acc[i][j] = zero;

  auto compute = [&](int buf) {
#pragma unroll
    for (int kh = 0; kh < 2; ++kh) {
      const _Float16* Ab = &Abuf[buf][kh * 4096];
      const _Float16* Bb = &Bbuf[buf][kh * 4096];
      v16h a0  = *(const v16h*)(Ab + (2 * waveM + 0) * 512 + lane * 16);
      v16h a1  = *(const v16h*)(Ab + (2 * waveM + 1) * 512 + lane * 16);
      v16h bf0 = *(const v16h*)(Bb + (4 * waveN + 0) * 512 + lane * 16);
      v16h bf1 = *(const v16h*)(Bb + (4 * waveN + 1) * 512 + lane * 16);
      v16h bf2 = *(const v16h*)(Bb + (4 * waveN + 2) * 512 + lane * 16);
      v16h bf3 = *(const v16h*)(Bb + (4 * waveN + 3) * 512 + lane * 16);
      acc[0][0] = __builtin_amdgcn_wmma_f32_16x16x32_f16(false, a0, false, bf0,
                                                         (short)0, acc[0][0], false, false);
      acc[0][1] = __builtin_amdgcn_wmma_f32_16x16x32_f16(false, a0, false, bf1,
                                                         (short)0, acc[0][1], false, false);
      acc[0][2] = __builtin_amdgcn_wmma_f32_16x16x32_f16(false, a0, false, bf2,
                                                         (short)0, acc[0][2], false, false);
      acc[0][3] = __builtin_amdgcn_wmma_f32_16x16x32_f16(false, a0, false, bf3,
                                                         (short)0, acc[0][3], false, false);
      acc[1][0] = __builtin_amdgcn_wmma_f32_16x16x32_f16(false, a1, false, bf0,
                                                         (short)0, acc[1][0], false, false);
      acc[1][1] = __builtin_amdgcn_wmma_f32_16x16x32_f16(false, a1, false, bf1,
                                                         (short)0, acc[1][1], false, false);
      acc[1][2] = __builtin_amdgcn_wmma_f32_16x16x32_f16(false, a1, false, bf2,
                                                         (short)0, acc[1][2], false, false);
      acc[1][3] = __builtin_amdgcn_wmma_f32_16x16x32_f16(false, a1, false, bf3,
                                                         (short)0, acc[1][3], false, false);
    }
  };

  stage(0, 0);
  __syncthreads();

  // K loop unrolled by 2 so LDS buffer bases are compile-time constants.
#pragma unroll 1
  for (int ks = 0; ks < KSTEPS; ks += 2) {
    stage(ks + 1, 1);                      // ks+1 <= KSTEPS-1 always (KSTEPS even)
    if (ks + 2 < KSTEPS)
      __builtin_prefetch(Wm + (tid >> 1) * NIN + (ks + 2) * TILE_K + (tid & 1) * 32,
                         0, 1);
    compute(0);
    __syncthreads();

    if (ks + 2 < KSTEPS) stage(ks + 2, 0);
    compute(1);
    __syncthreads();
  }

  // Epilogue: C/D 16x16 f32 layout: VGPR r -> M = r + 8*(lane>=16), N = lane&15
  float* __restrict__ Zb = Z + (size_t)b * NOUT * T_SMP;
#pragma unroll
  for (int i = 0; i < 2; ++i) {
    const int gm0 = mBase + waveM * 32 + i * 16 + 8 * (lane >> 4);
#pragma unroll
    for (int j = 0; j < 4; ++j) {
      const int col = nBase + waveN * 64 + j * 16 + (lane & 15);
      if (!GUARD || col < T_SMP) {
#pragma unroll
        for (int r = 0; r < 8; ++r)
          Zb[(size_t)(gm0 + r) * T_SMP + col] = acc[i][j][r];
      }
    }
  }
}

// ---------------------------------------------------------------------------
// Kernel 2: in-place causal conv with SRM alpha kernel along T.
// One block per (b, o) row; row cached in LDS so the overwrite is safe.
// ---------------------------------------------------------------------------
__global__ __launch_bounds__(256)
void slayer_psp_inplace(float* __restrict__ Z, SrmTaps taps)
{
  __shared__ float row[T_SMP];
  const int tid = threadIdx.x;
  float* zr = Z + (size_t)blockIdx.x * T_SMP;

  for (int t = tid; t < T_SMP; t += 256) row[t] = zr[t];
  __syncthreads();

  for (int t = tid; t < T_SMP; t += 256) {
    float acc = 0.0f;
    int kmax = taps.n - 1;
    if (kmax > t) kmax = t;
    for (int k = 0; k <= kmax; ++k) acc += taps.v[k] * row[t - k];
    zr[t] = acc;
  }
}

// ---------------------------------------------------------------------------
// Kernel 3: sequential threshold + refractory scan, one thread per row.
// new r[i] = old r[i+1] + s * ref[i]   (matches reference shift-then-inject)
// ---------------------------------------------------------------------------
__global__ __launch_bounds__(256)
void slayer_scan(const float* __restrict__ U, float* __restrict__ Sout, RefTaps ref)
{
  const int rowi = blockIdx.x * blockDim.x + threadIdx.x;
  if (rowi >= B_SZ * NOUT) return;
  const float* ur = U + (size_t)rowi * T_SMP;
  float* sr = Sout + (size_t)rowi * T_SMP;

  float r[16];
#pragma unroll
  for (int i = 0; i < 16; ++i) r[i] = 0.0f;

  for (int t = 0; t < T_SMP; ++t) {
    const float u  = ur[t] + r[0];
    const float sp = (u >= THETA) ? 1.0f : 0.0f;
#pragma unroll
    for (int i = 0; i < 15; ++i) r[i] = r[i + 1] + sp * ref.v[i];
    r[15] = sp * ref.v[15];
    sr[t] = sp;
  }
}

// ---------------------------------------------------------------------------
extern "C" void kernel_launch(void* const* d_in, const int* in_sizes, int n_in,
                              void* d_out, int out_size, void* d_ws, size_t ws_size,
                              hipStream_t stream)
{
  (void)in_sizes; (void)n_in; (void)out_size; (void)ws_size;

  const float* S = (const float*)d_in[0];   // (32, 2048, 500) f32
  const float* W = (const float*)d_in[1];   // (1024, 2048)    f32
  float* out = (float*)d_out;               // (32, 1024, 500) f32
  float* Z   = (float*)d_ws;                // scratch: 32*1024*500 f32 (64 MB)

  // SLAYER alpha kernels, identical truncation rule to the reference.
  SrmTaps srm;
  {
    int n = 0;
    for (int t = 0; t < 96; ++t) {
      double v = ((double)t / 10.0) * exp(1.0 - (double)t / 10.0);
      if (fabs(v) < 0.01 && (double)t > 10.0) break;
      srm.v[n++] = (float)v;
    }
    srm.n = n;                               // = 77
    for (int i = n; i < 96; ++i) srm.v[i] = 0.0f;
  }
  RefTaps ref;
  {
    for (int i = 0; i < 16; ++i) ref.v[i] = 0.0f;
    for (int t = 0; t < 16; ++t) {
      double v = (-2.0 * 10.0) * (double)t * exp(1.0 - (double)t);
      if (fabs(v) < 0.01 && (double)t > 1.0) break;  // Kr = 11
      ref.v[t] = (float)v;
    }
  }

  // Full tiles (no guards): cols [0, 384); guarded tile: cols [384, 512).
  const int fullTiles = T_SMP / TILE_N;                 // 3
  dim3 g0(fullTiles, NOUT / TILE_M, B_SZ);
  slayer_gemm_wmma<false><<<g0, dim3(256), 0, stream>>>(S, W, Z, 0);
  dim3 g1(1, NOUT / TILE_M, B_SZ);
  slayer_gemm_wmma<true><<<g1, dim3(256), 0, stream>>>(S, W, Z, fullTiles * TILE_N);

  slayer_psp_inplace<<<dim3(B_SZ * NOUT), dim3(256), 0, stream>>>(Z, srm);

  slayer_scan<<<dim3((B_SZ * NOUT + 255) / 256), dim3(256), 0, stream>>>(Z, out, ref);
}
